// RNA_LSTM_9620726743173
// MI455X (gfx1250) — compile-verified, matrix operand order checked by reading, not run on hardware
//
#include <hip/hip_runtime.h>

// ---------------------------------------------------------------------------
// MI455X (gfx1250) fused bidirectional-LSTM.
//
// ~70 GFLOP total; the inter-layer activation (64 MB f16) is L2-resident
// (192 MB @ effectively L2 bandwidth), so the limiter is the T=512 sequential
// recurrence. Per step, one fused WMMA GEMM:
//   [16 x (I+H)] x [(I+H) x 256] -> [16 x 256] gate pre-activations
// with all weight B-fragments resident in VGPRs across all 512 steps; only
// h (2 KB) and the x tile (<=4 KB) round-trip through LDS.
//
// Wave mapping (wave32, 4 waves, 16 batch rows / workgroup): wave w owns
// h-columns [16w,16w+16) => gate N-tiles {w, w+4, w+8, w+12} (i,f,g,o), so
// gate nonlinearities and the c/h update are wave-local fragment ops.
// gridDim.y = direction (0=fwd, 1=bwd) -> both directions run concurrently.
// ---------------------------------------------------------------------------

typedef __attribute__((ext_vector_type(16))) _Float16 v16h;
typedef __attribute__((ext_vector_type(8)))  _Float16 v8h;
typedef __attribute__((ext_vector_type(8)))  float    v8f;

constexpr int Bsz = 512;   // batch
constexpr int Tsz = 512;   // sequence length
constexpr int Hsz = 64;    // hidden
constexpr int NG  = 256;   // 4*H gates

// Gate activations on the sequential critical path: use raw v_rcp_f32
// (~1 ulp) instead of IEEE division (div_scale/div_fmas expansion).
__device__ __forceinline__ float sigm(float x) {
  return __builtin_amdgcn_rcpf(1.0f + __expf(-x));
}
__device__ __forceinline__ float tanh_fast(float x) {
  return 1.0f - 2.0f * __builtin_amdgcn_rcpf(1.0f + __expf(2.0f * x));
}

// 16-bit A/B fragment element->K map (ISA 7.12.2, 16-bit 16x32 tile):
//   lane group (lane>=16) adds +8, element e: K = 16*(e/8) + (e&7)
__device__ __forceinline__ int frag_k(int e, int lane, int k_off) {
  return k_off + ((lane >> 4) << 3) + ((e >> 3) << 4) + (e & 7);
}

// A fragment from row-major LDS buffer [16][STRIDE] (row = batch M = lane&15).
// The 8-element contiguous runs vectorize into ds_load_b128 pairs.
template<int STRIDE>
__device__ __forceinline__ v16h load_fragA(const _Float16* buf, int k_off, int lane) {
  const int r = lane & 15;
  v16h f;
#pragma unroll
  for (int e = 0; e < 16; ++e)
    f[e] = buf[r * STRIDE + frag_k(e, lane, k_off)];
  return f;
}

// B fragment gathered once from global fp32 weight W[NG][ldw] (row-major,
// gate-major as in PyTorch): B[k][n] = W[n][k], zero-padded past kreal.
__device__ __forceinline__ v16h load_fragB_global(const float* w, int ldw, int kreal,
                                                  int k_off, int n_off, int lane) {
  const int n = n_off + (lane & 15);
  v16h f;
#pragma unroll
  for (int e = 0; e < 16; ++e) {
    int k = frag_k(e, lane, k_off);
    f[e] = (k < kreal) ? (_Float16)w[n * ldw + k] : (_Float16)0.0f;
  }
  return f;
}

// KI   : padded K of the x-projection (multiple of 32): 32 (layer0) / 128 (layer1)
// KREAL: true input width: 4 (layer0) / 128 (layer1)
// IS_L1: layer-1 fuses the T-mean pooling instead of writing h per step
template<int KI, int KREAL, bool IS_L1>
__global__ __launch_bounds__(128)
void lstm_layer_kernel(const float*    __restrict__ x0,     // [B,T,4]   (layer0)
                       const _Float16* __restrict__ hin,    // [B,T,128] (layer1)
                       const float*    __restrict__ Wih_f,  // [256,KREAL]
                       const float*    __restrict__ Whh_f,  // [256,64]
                       const float*    __restrict__ b_f,    // [256]
                       const float*    __restrict__ Wih_b,
                       const float*    __restrict__ Whh_b,
                       const float*    __restrict__ b_b,
                       _Float16*       __restrict__ hout,   // [B,T,128] (layer0)
                       float*          __restrict__ pooled) // [B,128]   (layer1)
{
  constexpr int NKX = KI / 32;       // K-chunks of the x projection
  __shared__ _Float16 sH[16 * Hsz];  // current h, row-major [batch16][64]
  __shared__ _Float16 sX[16 * KI];   // current x tile, row-major

  const int tid   = threadIdx.x;
  const int lane  = tid & 31;
  const int cg    = tid >> 5;        // wave id == 16-column group of h
  const int bbase = blockIdx.x * 16;
  const int dir   = blockIdx.y;      // 0 = forward, 1 = backward (reversed scan)

  const float* Wih  = dir ? Wih_b : Wih_f;
  const float* Whh  = dir ? Whh_b : Whh_f;
  const float* bias = dir ? b_b   : b_f;

  for (int idx = tid; idx < 16 * Hsz; idx += 128) sH[idx] = (_Float16)0.0f;
  for (int idx = tid; idx < 16 * KI;  idx += 128) sX[idx] = (_Float16)0.0f;

  // ---- loop-invariant weight B-fragments: resident in VGPRs for all 512 steps
  v16h bX[NKX][4];   // x-projection weights  (gate tile j at n = 16*cg + 64*j)
  v16h bH[2][4];     // recurrent weights (K=64 -> 2 chunks)
  float bval[4];     // bias per owned gate column
#pragma unroll
  for (int j = 0; j < 4; ++j) {
    const int n_off = 16 * cg + 64 * j;
#pragma unroll
    for (int kc = 0; kc < NKX; ++kc)
      bX[kc][j] = load_fragB_global(Wih, KREAL, KREAL, 32 * kc, n_off, lane);
#pragma unroll
    for (int kc = 0; kc < 2; ++kc)
      bH[kc][j] = load_fragB_global(Whh, Hsz, Hsz, 32 * kc, n_off, lane);
    bval[j] = bias[n_off + (lane & 15)];
  }

  v8f c, hsum;
#pragma unroll
  for (int v = 0; v < 8; ++v) { c[v] = 0.0f; hsum[v] = 0.0f; }

  const int col   = (cg << 4) + (lane & 15);   // owned h column
  const int rbase = (lane >> 4) << 3;          // D-fragment row base

  __syncthreads();

  for (int t = 0; t < Tsz; ++t) {
    const int tt = dir ? (Tsz - 1 - t) : t;

    // ---- stage x_t tile into LDS (vectorized: b128 global, b128/b64 LDS)
    if (!IS_L1) {
      if (tid < 16) {                          // one float4 row-load per thread
        const float4 xv = *(const float4*)&x0[((size_t)(bbase + tid) * Tsz + tt) * 4];
        union { _Float16 h[4]; unsigned long long u; } pk;
        pk.h[0] = (_Float16)xv.x; pk.h[1] = (_Float16)xv.y;
        pk.h[2] = (_Float16)xv.z; pk.h[3] = (_Float16)xv.w;
        *(unsigned long long*)&sX[tid * KI] = pk.u;   // cols 4..31 stay 0
      }
    } else {
      // 16 rows x 256 B = 256 v8h chunks over 128 threads (2 per thread)
#pragma unroll
      for (int it = 0; it < 2; ++it) {
        int idx = tid + it * 128;
        int r = idx >> 4, ch = idx & 15;
        ((v8h*)sX)[r * 16 + ch] =
            *(const v8h*)&hin[((size_t)(bbase + r) * Tsz + tt) * 128 + ch * 8];
      }
      if (t + 1 < Tsz) {                       // prefetch next step's tile (L2-resident)
        int tn = dir ? (Tsz - 2 - t) : (t + 1);
        __builtin_prefetch(&hin[((size_t)(bbase + (tid >> 3)) * Tsz + tn) * 128 + (tid & 7) * 16], 0, 1);
      }
    }
    __syncthreads();   // sX ready; previous-step sH writes visible

    // ---- A fragments for this step
    v16h aX[NKX], aH[2];
#pragma unroll
    for (int kc = 0; kc < NKX; ++kc) aX[kc] = load_fragA<KI>(sX, 32 * kc, lane);
#pragma unroll
    for (int kc = 0; kc < 2; ++kc)   aH[kc] = load_fragA<Hsz>(sH, 32 * kc, lane);
    __syncthreads();   // all reads of sH/sX done before they are overwritten

    // ---- g = bias + x_t @ Wih^T + h @ Whh^T  (f32 accumulate)
    v8f acc[4];
#pragma unroll
    for (int j = 0; j < 4; ++j) {
#pragma unroll
      for (int v = 0; v < 8; ++v) acc[j][v] = bval[j];
#pragma unroll
      for (int kc = 0; kc < NKX; ++kc)
        acc[j] = __builtin_amdgcn_wmma_f32_16x16x32_f16(false, aX[kc], false, bX[kc][j],
                                                        (short)0, acc[j], false, false);
#pragma unroll
      for (int kc = 0; kc < 2; ++kc)
        acc[j] = __builtin_amdgcn_wmma_f32_16x16x32_f16(false, aH[kc], false, bH[kc][j],
                                                        (short)0, acc[j], false, false);
    }

    // ---- gates + state update (wave-local: identical D layouts elementwise)
    v8f hnew;
#pragma unroll
    for (int v = 0; v < 8; ++v) {
      float ig = sigm(acc[0][v]);
      float fg = sigm(acc[1][v]);
      float gg = tanh_fast(acc[2][v]);
      float og = sigm(acc[3][v]);
      float cc = fg * c[v] + ig * gg;
      c[v]     = cc;
      hnew[v]  = og * tanh_fast(cc);
    }

#pragma unroll
    for (int v = 0; v < 8; ++v)
      sH[(rbase + v) * Hsz + col] = (_Float16)hnew[v];

    if (!IS_L1) {
#pragma unroll
      for (int v = 0; v < 8; ++v)
        hout[((size_t)(bbase + rbase + v) * Tsz + tt) * 128 + dir * Hsz + col] = (_Float16)hnew[v];
    } else {
#pragma unroll
      for (int v = 0; v < 8; ++v) hsum[v] += hnew[v];
    }
  }

  if (IS_L1) {   // fused mean over T; fwd/bwd write disjoint column halves
#pragma unroll
    for (int v = 0; v < 8; ++v)
      pooled[(size_t)(bbase + rbase + v) * 128 + dir * Hsz + col] = hsum[v] * (1.0f / Tsz);
  }
}

// out[b] = pooled[b,:] . fc_w + fc_b   (one wave32 per batch row)
__global__ __launch_bounds__(128)
void fc_pool_kernel(const float* __restrict__ pooled,
                    const float* __restrict__ fcw,
                    const float* __restrict__ fcb,
                    float* __restrict__ out)
{
  int b    = blockIdx.x * 4 + (threadIdx.x >> 5);
  int lane = threadIdx.x & 31;
  float s = 0.0f;
  for (int ci = lane; ci < 128; ci += 32) s += pooled[(size_t)b * 128 + ci] * fcw[ci];
#pragma unroll
  for (int off = 16; off > 0; off >>= 1) s += __shfl_down(s, off, 32);
  if (lane == 0) out[b] = s + fcb[0];
}

extern "C" void kernel_launch(void* const* d_in, const int* in_sizes, int n_in,
                              void* d_out, int out_size, void* d_ws, size_t ws_size,
                              hipStream_t stream) {
  (void)in_sizes; (void)n_in; (void)out_size; (void)ws_size;
  const float* x     = (const float*)d_in[0];
  const float* Wih0f = (const float*)d_in[1];
  const float* Whh0f = (const float*)d_in[2];
  const float* b0f   = (const float*)d_in[3];
  const float* Wih0b = (const float*)d_in[4];
  const float* Whh0b = (const float*)d_in[5];
  const float* b0b   = (const float*)d_in[6];
  const float* Wih1f = (const float*)d_in[7];
  const float* Whh1f = (const float*)d_in[8];
  const float* b1f   = (const float*)d_in[9];
  const float* Wih1b = (const float*)d_in[10];
  const float* Whh1b = (const float*)d_in[11];
  const float* b1b   = (const float*)d_in[12];
  const float* fcw   = (const float*)d_in[13];
  const float* fcb   = (const float*)d_in[14];
  float* out = (float*)d_out;

  // Workspace: h0 activations [B,T,128] f16 (64 MB, L2-resident), then pooled [B,128] f32.
  _Float16* h0  = (_Float16*)d_ws;
  float* pooled = (float*)((char*)d_ws + (size_t)Bsz * Tsz * 128 * sizeof(_Float16));

  dim3 grid(Bsz / 16, 2), blk(128);   // grid.y = direction (fwd/bwd concurrent)
  lstm_layer_kernel<32, 4, false><<<grid, blk, 0, stream>>>(
      x, nullptr, Wih0f, Whh0f, b0f, Wih0b, Whh0b, b0b, h0, nullptr);
  lstm_layer_kernel<128, 128, true><<<grid, blk, 0, stream>>>(
      nullptr, h0, Wih1f, Whh1f, b1f, Wih1b, Whh1b, b1b, nullptr, pooled);
  fc_pool_kernel<<<dim3(Bsz / 4), blk, 0, stream>>>(pooled, fcw, fcb, out);
}
